// MultiHeadAttentionBlock_1606317768891
// MI455X (gfx1250) — compile-verified, matrix operand order checked by reading, not run on hardware
//
#include <hip/hip_runtime.h>

typedef __attribute__((ext_vector_type(16))) __bf16 v16bf;
typedef __attribute__((ext_vector_type(8)))  float  v8f;
typedef unsigned short ushort_t;

typedef __attribute__((ext_vector_type(4))) unsigned int u32x4;
typedef __attribute__((ext_vector_type(8))) int          i32x8;
typedef __attribute__((ext_vector_type(4))) int          i32x4;

// ---- Tensor Data Mover feature detection (with stderr diagnostics) --------
#if defined(__gfx1250__)
#if !__has_builtin(__builtin_amdgcn_tensor_load_to_lds)
#warning "TDM probe: __builtin_amdgcn_tensor_load_to_lds NOT available"
#endif
#if !__has_builtin(__builtin_amdgcn_s_wait_tensorcnt)
#warning "TDM probe: __builtin_amdgcn_s_wait_tensorcnt NOT available"
#endif
#if __has_include(<hip/amd_detail/amd_gfx1250_TDM.h>)
#warning "TDM probe: amd_gfx1250_TDM.h present (therock / 6-arg builtin)"
#endif
#endif

#if defined(__gfx1250__) && __has_builtin(__builtin_amdgcn_tensor_load_to_lds) && \
    __has_builtin(__builtin_amdgcn_s_wait_tensorcnt)
#define USE_TDM 1
#if __has_include(<hip/amd_detail/amd_gfx1250_TDM.h>)
#define TDM_ARITY6 1  // therock clang-23: extra i32x8 operand
#else
#define TDM_ARITY6 0  // ROCm 7.2 clang-22: 5-arg form
#endif
#else
#define USE_TDM 0
#endif

constexpr int B = 2, S = 2048, D = 1024, H = 16, DK = 64;

union Frag {
  v16bf v;
  uint2 d2[4];
  uint4 d4[2];
  ushort_t s[16];
};
union Pack4 { ushort_t s[4]; uint2 u; };
union Pack8 { ushort_t s[8]; uint4 u; };

// Native f32->bf16 (RNE) conversion; backend selects the hardware cvt.
__device__ __forceinline__ ushort_t f2bf(float x) {
  union { __bf16 h; ushort_t s; } c;
  c.h = (__bf16)x;
  return c.s;
}

__device__ __forceinline__ v8f wmma_bf16(v16bf a, v16bf b, v8f c) {
  // emits v_wmma_f32_16x16x32_bf16 (8-arg form, c_mod=0, no reuse hints)
  return __builtin_amdgcn_wmma_f32_16x16x32_bf16(false, a, false, b, (short)0, c,
                                                 false, false);
}

// Load a 16x32 bf16 fragment from an LDS tile stored row-major [outer][inner].
// A-matrix 16x32 layout (ISA 7.12.2): lane L holds row outer0+(L&15);
// elements 0-7 -> inner = inner0 + (L>>4)*8 + e ; elements 8-15 -> +16.
// Same gather serves B when the tile is stored [N][K].
__device__ __forceinline__ v16bf load_frag_lds(const ushort_t* t, int outer0,
                                               int inner0, int pitch, int lane) {
  Frag f;
  const ushort_t* p = t + (outer0 + (lane & 15)) * pitch + inner0 + (lane >> 4) * 8;
  f.d2[0] = *(const uint2*)(p + 0);
  f.d2[1] = *(const uint2*)(p + 4);
  f.d2[2] = *(const uint2*)(p + 16);
  f.d2[3] = *(const uint2*)(p + 20);
  return f.v;
}

// ---------------------------------------------------------------------------
// Kernel 1: y = x @ W^T for q/k/v (blockIdx.z selects tensor).
// 128x128 output tile per 256-thread block (8 waves, 4Mx2N, 32x64 each).
// f32 inputs converted to bf16 into LDS; output stored bf16 in [B,H,S,DK].
// ---------------------------------------------------------------------------
constexpr int PITCH1 = 36; // ushorts; 72B rows -> conflict-free b64 frag reads

__global__ __launch_bounds__(256) void qkv_proj_kernel(
    const float* __restrict__ xq, const float* __restrict__ xk,
    const float* __restrict__ xv, const float* __restrict__ wq,
    const float* __restrict__ wk, const float* __restrict__ wv,
    ushort_t* __restrict__ oq, ushort_t* __restrict__ ok,
    ushort_t* __restrict__ ov) {
  const int nb = blockIdx.x, mb = blockIdx.y, t = blockIdx.z;
  const float* x = (t == 0) ? xq : (t == 1) ? xk : xv;
  const float* w = (t == 0) ? wq : (t == 1) ? wk : wv;
  ushort_t* o = (t == 0) ? oq : (t == 1) ? ok : ov;

  __shared__ __align__(16) ushort_t sA[128 * PITCH1];
  __shared__ __align__(16) ushort_t sB[128 * PITCH1];

  const int tid = threadIdx.x;
  const int lane = tid & 31, wave = tid >> 5;
  const int wm = wave & 3, wn = wave >> 2;
  const int half = lane >> 4, l15 = lane & 15;

  const v8f vzero = {0.f, 0.f, 0.f, 0.f, 0.f, 0.f, 0.f, 0.f};
  v8f acc[2][4];
#pragma unroll
  for (int i = 0; i < 2; ++i)
#pragma unroll
    for (int j = 0; j < 4; ++j) acc[i][j] = vzero;

  const long xrow0 = (long)(mb * 128) * D;
  const long wrow0 = (long)(nb * 128) * D;

  for (int kt = 0; kt < D / 32; ++kt) {
    const int k0 = kt * 32;
    __syncthreads();
    // 128x32 f32 -> bf16 staging for both operands (1024 float4 chunks each)
#pragma unroll
    for (int c = 0; c < 4; ++c) {
      int idx = c * 256 + tid;
      int row = idx >> 3, col = (idx & 7) * 4;
      float4 fx = *(const float4*)(x + xrow0 + (long)row * D + k0 + col);
      float4 fw = *(const float4*)(w + wrow0 + (long)row * D + k0 + col);
      Pack4 pa, pb;
      pa.s[0] = f2bf(fx.x); pa.s[1] = f2bf(fx.y);
      pa.s[2] = f2bf(fx.z); pa.s[3] = f2bf(fx.w);
      pb.s[0] = f2bf(fw.x); pb.s[1] = f2bf(fw.y);
      pb.s[2] = f2bf(fw.z); pb.s[3] = f2bf(fw.w);
      *(uint2*)(sA + row * PITCH1 + col) = pa.u;
      *(uint2*)(sB + row * PITCH1 + col) = pb.u;
    }
    if (kt + 1 < D / 32) {  // global_prefetch_b8 for next K block
      int prow = tid & 127;
      __builtin_prefetch(x + xrow0 + (long)prow * D + k0 + 32, 0, 1);
      __builtin_prefetch(w + wrow0 + (long)prow * D + k0 + 32, 0, 1);
    }
    __syncthreads();

    v16bf af[2], bfr[4];
#pragma unroll
    for (int i = 0; i < 2; ++i)
      af[i] = load_frag_lds(sA, wm * 32 + i * 16, 0, PITCH1, lane);
#pragma unroll
    for (int j = 0; j < 4; ++j)
      bfr[j] = load_frag_lds(sB, wn * 64 + j * 16, 0, PITCH1, lane);
#pragma unroll
    for (int i = 0; i < 2; ++i)
#pragma unroll
      for (int j = 0; j < 4; ++j) acc[i][j] = wmma_bf16(af[i], bfr[j], acc[i][j]);
  }

  // D-layout (ISA 7.12.2): row = 8*half + vgpr, col = lane%16. Store [B,H,S,DK].
#pragma unroll
  for (int i = 0; i < 2; ++i)
#pragma unroll
    for (int j = 0; j < 4; ++j)
#pragma unroll
      for (int r = 0; r < 8; ++r) {
        int m = mb * 128 + wm * 32 + i * 16 + half * 8 + r;
        int oc = nb * 128 + wn * 64 + j * 16 + l15;
        int bb = m / S, s = m % S;
        int hh = oc / DK, dk = oc % DK;
        long addr = (((long)(bb * H + hh) * S) + s) * DK + dk;
        o[addr] = f2bf(acc[i][j][r]);
      }
}

// ---------------------------------------------------------------------------
// Kernel 2: flash-attention over causal key blocks of 128.
// Block = 128 query rows x one (b,h); 8 waves of 16 rows each.
// Q fragments register-resident; K tile [key][dk] staged by the Tensor Data
// Mover when available (pad feature reproduces the pitch-72 layout); V tile
// transposed [dk][key]; P routed through LDS (reusing the K region) for P@V.
// ---------------------------------------------------------------------------
constexpr int KP = 72;   // K tile pitch (ushorts), 144B rows = 32+4 DWORDs
constexpr int PP = 136;  // P / Vt pitch (ushorts), 272B rows, 16B aligned

__global__ __launch_bounds__(256) void attention_kernel(
    const ushort_t* __restrict__ Q, const ushort_t* __restrict__ K,
    const ushort_t* __restrict__ V, float* __restrict__ out) {
  const int mb = blockIdx.x;  // 16 query blocks
  const int bh = blockIdx.y;  // 32 (b,h) pairs
  const int b = bh / H, h = bh % H;

  __shared__ __align__(16) ushort_t sKP[128 * PP];  // K tile, then reused as P
  __shared__ __align__(16) ushort_t sVt[64 * PP];   // V transposed [dk][key]

  const int tid = threadIdx.x, lane = tid & 31, wave = tid >> 5;
  const int half = lane >> 4, l15 = lane & 15;
  const int m0 = wave * 16;

  const ushort_t* Qb = Q + ((long)bh * S + mb * 128) * DK;
  const ushort_t* Kb0 = K + (long)bh * S * DK;
  const ushort_t* Vb0 = V + (long)bh * S * DK;

  // Resident Q A-fragments: 2 k-steps covering DK=64 (aligned 16B chunks).
  v16bf qf[2];
#pragma unroll
  for (int ks = 0; ks < 2; ++ks) {
    Frag f;
    const ushort_t* p = Qb + (m0 + l15) * DK + ks * 32 + half * 8;
    f.d4[0] = *(const uint4*)(p);
    f.d4[1] = *(const uint4*)(p + 16);
    qf[ks] = f.v;
  }

  const v8f vzero = {0.f, 0.f, 0.f, 0.f, 0.f, 0.f, 0.f, 0.f};
  v8f Ot[4];
  float mrow[8], lrow[8];
#pragma unroll
  for (int t = 0; t < 4; ++t) Ot[t] = vzero;
#pragma unroll
  for (int r = 0; r < 8; ++r) { mrow[r] = -3.0e38f; lrow[r] = 0.0f; }

  for (int kb = 0; kb <= mb; ++kb) {
    const ushort_t* Kb = Kb0 + (long)kb * 128 * DK;
    const ushort_t* Vb = Vb0 + (long)kb * 128 * DK;
    __syncthreads();

#if USE_TDM
    // --- K tile via Tensor Data Mover: 128x64 bf16, LDS rows padded 32+4 DW.
    if (wave == 0) {
      unsigned long long ga = (unsigned long long)Kb;
      unsigned lds_off = (unsigned)(unsigned long long)(&sKP[0]);
      u32x4 g0;
      g0[0] = 1u;                                    // count=1, user mode
      g0[1] = lds_off;                               // D#.lds_addr (bytes)
      g0[2] = (unsigned)(ga & 0xFFFFFFFFu);          // global_addr[31:0]
      g0[3] = (unsigned)((ga >> 32) & 0x01FFFFFFu)   // global_addr[56:32]
              | 0x80000000u;                         // type=2 ("image")
      i32x8 g1;
      g1[0] = (int)((1u << 16)      // data_size = 2 bytes
                    | (1u << 20)    // pad_enable
                    | (4u << 22)    // pad_interval: 32 DWORDs (one 64-elt row)
                    | (3u << 25));  // pad_amount: 4 DWORDs (-> pitch 72)
      g1[1] = (int)(64u << 16);     // tensor_dim0 = 64
      g1[2] = (int)(128u << 16);    // tensor_dim1 = 128
      g1[3] = (int)(64u << 16);     // tile_dim0 = 64
      g1[4] = 128;                  // tile_dim1 = 128, tile_dim2 = 0
      g1[5] = 64;                   // tensor_dim0_stride = 64
      g1[6] = 0;
      g1[7] = 0;
      i32x4 gz = {0, 0, 0, 0};      // groups 2/3 unused (2-D tensor)
#if TDM_ARITY6
      i32x8 gz8 = {0, 0, 0, 0, 0, 0, 0, 0};
      __builtin_amdgcn_tensor_load_to_lds(g0, g1, gz, gz, gz8, 0);
#else
      __builtin_amdgcn_tensor_load_to_lds(g0, g1, gz, gz, 0);
#endif
    }
    // --- V tile (needs transpose -> manual staging), all waves.
#pragma unroll
    for (int c = 0; c < 4; ++c) {
      int idx = c * 256 + tid;
      int row = idx >> 3, c8 = idx & 7;
      Pack8 vv; vv.u = *(const uint4*)(Vb + row * DK + c8 * 8);
#pragma unroll
      for (int e = 0; e < 8; ++e) sVt[(c8 * 8 + e) * PP + row] = vv.s[e];
    }
    if (wave == 0) __builtin_amdgcn_s_wait_tensorcnt(0);
#else
    // Manual staging: K [key][dk] and V transposed [dk][key].
#pragma unroll
    for (int c = 0; c < 4; ++c) {
      int idx = c * 256 + tid;
      int row = idx >> 3, c8 = idx & 7;
      uint4 kq = *(const uint4*)(Kb + row * DK + c8 * 8);
      *(uint4*)(sKP + row * KP + c8 * 8) = kq;
      Pack8 vv; vv.u = *(const uint4*)(Vb + row * DK + c8 * 8);
#pragma unroll
      for (int e = 0; e < 8; ++e) sVt[(c8 * 8 + e) * PP + row] = vv.s[e];
    }
#endif
    if (kb < mb) {  // prefetch next K/V blocks
      __builtin_prefetch(Kb + 128 * DK + tid * 32, 0, 1);
      __builtin_prefetch(Vb + 128 * DK + tid * 32, 0, 1);
    }
    __syncthreads();

    // Scores: S = Q @ K^T  (B gathered from [N=key][K=dk] tile)
    v8f sc[8];
#pragma unroll
    for (int j = 0; j < 8; ++j) sc[j] = vzero;
#pragma unroll
    for (int ks = 0; ks < 2; ++ks)
#pragma unroll
      for (int j = 0; j < 8; ++j) {
        v16bf kf = load_frag_lds(sKP, j * 16, ks * 32, KP, lane);
        sc[j] = wmma_bf16(qf[ks], kf, sc[j]);
      }
    __syncthreads();  // everyone done reading K before P overwrites the region

    // Online softmax. Per lane: row = m0 + 8*half + r, key = kb*128 + 16j + l15.
    float alpha[8];
#pragma unroll
    for (int r = 0; r < 8; ++r) {
      int qrow = mb * 128 + m0 + half * 8 + r;
      float mt = -3.0e38f;
#pragma unroll
      for (int j = 0; j < 8; ++j) {
        int kc = kb * 128 + j * 16 + l15;
        float sv = sc[j][r] * 0.125f;          // 1/sqrt(DK)
        sv = (kc <= qrow) ? sv : -1.0e9f;      // causal mask
        sc[j][r] = sv;
        mt = fmaxf(mt, sv);
      }
      mt = fmaxf(mt, __shfl_xor(mt, 1, 32));
      mt = fmaxf(mt, __shfl_xor(mt, 2, 32));
      mt = fmaxf(mt, __shfl_xor(mt, 4, 32));
      mt = fmaxf(mt, __shfl_xor(mt, 8, 32));
      float mn = fmaxf(mrow[r], mt);
      alpha[r] = __expf(mrow[r] - mn);
      float rs = 0.0f;
#pragma unroll
      for (int j = 0; j < 8; ++j) {
        float p = __expf(sc[j][r] - mn);
        sc[j][r] = p;
        rs += p;
      }
      rs += __shfl_xor(rs, 1, 32);
      rs += __shfl_xor(rs, 2, 32);
      rs += __shfl_xor(rs, 4, 32);
      rs += __shfl_xor(rs, 8, 32);
      lrow[r] = lrow[r] * alpha[r] + rs;
      mrow[r] = mn;
    }
#pragma unroll
    for (int t = 0; t < 4; ++t)
#pragma unroll
      for (int r = 0; r < 8; ++r) Ot[t][r] *= alpha[r];

    // P (D-layout) -> bf16 into this wave's own 16 rows of sKP (A-layout tile).
    // Same-wave DS ops are in-order, so no barrier needed before re-reading.
#pragma unroll
    for (int j = 0; j < 8; ++j)
#pragma unroll
      for (int r = 0; r < 8; ++r)
        sKP[(m0 + half * 8 + r) * PP + j * 16 + l15] = f2bf(sc[j][r]);

    // O += P @ V  (A from sKP rows m0..m0+15, B from transposed V tile)
#pragma unroll
    for (int ks = 0; ks < 4; ++ks) {
      v16bf pf = load_frag_lds(sKP, m0, ks * 32, PP, lane);
#pragma unroll
      for (int t = 0; t < 4; ++t) {
        v16bf vf = load_frag_lds(sVt, t * 16, ks * 32, PP, lane);
        Ot[t] = wmma_bf16(pf, vf, Ot[t]);
      }
    }
  }

  // Normalize and store f32 to [B,S,D] with D = H*DK.
#pragma unroll
  for (int t = 0; t < 4; ++t)
#pragma unroll
    for (int r = 0; r < 8; ++r) {
      int s = mb * 128 + m0 + half * 8 + r;
      int col = h * DK + t * 16 + l15;
      out[((long)(b * S + s)) * D + col] = Ot[t][r] / lrow[r];
    }
}

extern "C" void kernel_launch(void* const* d_in, const int* in_sizes, int n_in,
                              void* d_out, int out_size, void* d_ws,
                              size_t ws_size, hipStream_t stream) {
  const float* q  = (const float*)d_in[0];
  const float* k  = (const float*)d_in[1];
  const float* v  = (const float*)d_in[2];
  const float* wq = (const float*)d_in[3];
  const float* wk = (const float*)d_in[4];
  const float* wv = (const float*)d_in[5];
  // d_in[6] = causal mask (recomputed analytically in-kernel)

  ushort_t* wsQ = (ushort_t*)d_ws;          // bf16 [B,H,S,DK]
  ushort_t* wsK = wsQ + (long)B * S * D;    // 8 MB each, 24 MB total
  ushort_t* wsV = wsK + (long)B * S * D;
  float* out = (float*)d_out;

  dim3 g1(D / 128, (B * S) / 128, 3);       // (8, 32, 3)
  qkv_proj_kernel<<<g1, 256, 0, stream>>>(q, k, v, wq, wk, wv, wsQ, wsK, wsV);

  dim3 g2(S / 128, B * H, 1);               // (16, 32)
  attention_kernel<<<g2, 256, 0, stream>>>(wsQ, wsK, wsV, out);
}
